// AttentionModel_39848706573050
// MI455X (gfx1250) — matrix operand order repaired and checked by reading, not compiled, not run on hardware
//
#include <hip/hip_runtime.h>

// Problem constants (match reference)
#define B_  16
#define S_  2048
#define E_  128
#define SCALE_ 0.088388347648318447f   // 128^-0.5

// LDS layout per wave: 16 rows x (2048 + 8 pad) bf16 scores + 16 f32 stats
#define ROW_STRIDE   2056                      // bf16 elements, padded -> conflict-free col reads
#define SCORES_BYTES (16 * ROW_STRIDE * 2)     // 65792
#define REGION       (SCORES_BYTES + 64)       // 65856 (16-byte aligned)

typedef __attribute__((ext_vector_type(16))) __bf16 v16bf;
typedef __attribute__((ext_vector_type(2)))  __bf16 v2bf;
typedef __attribute__((ext_vector_type(8)))  float  v8f;

union FragBF {
    v16bf          v;
    unsigned short u[16];
    unsigned int   w[8];
    uint4          q[2];
};

// Native conversions: plain __bf16 casts lower to v_cvt_*bf16_f32 on gfx1250
__device__ __forceinline__ unsigned short f2bf(float f) {
    __bf16 h = (__bf16)f;
    return __builtin_bit_cast(unsigned short, h);
}
__device__ __forceinline__ unsigned int f2bf_pk(float a, float b) {
    v2bf h;
    h.x = (__bf16)a;
    h.y = (__bf16)b;
    return __builtin_bit_cast(unsigned int, h);
}
__device__ __forceinline__ float bf2f(unsigned int us) {
    return __uint_as_float(us << 16);
}

// ---------------------------------------------------------------------------
// Kernel 1: K (fp32 [B,S,E]) -> bf16 [B,S,E];  V (fp32 [B,S,E]) -> bf16 [B,E,S]
// ---------------------------------------------------------------------------
__global__ __launch_bounds__(256) void cvt_kv_kernel(
    const float* __restrict__ k, const float* __restrict__ v,
    unsigned short* __restrict__ kbf, unsigned short* __restrict__ vtbf)
{
    unsigned int i = blockIdx.x * 256u + threadIdx.x;   // 0 .. B*S*E-1 (4,194,304)
    kbf[i] = f2bf(k[i]);
    unsigned int e = i & (E_ - 1);
    unsigned int s = (i >> 7) & (S_ - 1);
    unsigned int b = i >> 18;                           // / (S_*E_)
    vtbf[((size_t)b << 18) + ((size_t)e << 11) + s] = f2bf(v[i]);
}

// ---------------------------------------------------------------------------
// Kernel 2: attention. 4 waves per block, one 16-row query tile per wave.
// ---------------------------------------------------------------------------
__global__ __launch_bounds__(128) void attn_kernel(
    const float* __restrict__ q,
    const unsigned short* __restrict__ kbf,     // [B,S,E] bf16
    const unsigned short* __restrict__ vtbf,    // [B,E,S] bf16
    float* __restrict__ out)
{
    extern __shared__ unsigned char smem_raw[];

    const int lane = threadIdx.x & 31;
    const int wave = threadIdx.x >> 5;
    const int tile = blockIdx.x * 4 + wave;     // 0 .. 2047
    const int b    = tile >> 7;                 // batch
    const int qb   = (tile & 127) << 4;         // query row base
    const int m    = lane & 15;                 // row-in-tile (A) / N (B,C,D)
    const int half = lane >> 4;

    unsigned short* sc    = (unsigned short*)(smem_raw + (size_t)wave * REGION);
    float*          stats = (float*)(smem_raw + (size_t)wave * REGION + SCORES_BYTES);

    // ---- Load Q A-fragments (16x32 bf16 each, K-slices 0,32,64,96) ----
    FragBF qa[4];
    const float* qrow = q + ((size_t)(b * S_ + qb + m)) * E_;
    #pragma unroll
    for (int ks4 = 0; ks4 < 4; ++ks4) {
        #pragma unroll
        for (int v = 0; v < 8; ++v) {
            const int kofs = 16 * half + 2 * v;
            const float2 f = *(const float2*)(qrow + ks4 * 32 + kofs);
            qa[ks4].w[v] = f2bf_pk(f.x, f.y);
        }
    }

    // ---- Phase 1: scores = scale * Q K^T, bf16 into LDS ----
    const unsigned short* kbase = kbf + ((size_t)b * S_) * E_;
    for (int kt = 0; kt < 128; ++kt) {
        const unsigned short* krow = kbase + ((size_t)(kt * 16 + m)) * E_ + 16 * half;
        __builtin_prefetch(krow + E_ * 16, 0, 1);    // next key tile -> global_prefetch
        v8f c = {0.f, 0.f, 0.f, 0.f, 0.f, 0.f, 0.f, 0.f};
        #pragma unroll
        for (int ks4 = 0; ks4 < 4; ++ks4) {
            FragBF kb_;
            const uint4* p = (const uint4*)(krow + ks4 * 32);
            kb_.q[0] = p[0];
            kb_.q[1] = p[1];
            c = __builtin_amdgcn_wmma_f32_16x16x32_bf16(
                    false, qa[ks4].v, false, kb_.v, (short)0, c, false, false);
        }
        #pragma unroll
        for (int r = 0; r < 8; ++r)
            sc[(r + 8 * half) * ROW_STRIDE + kt * 16 + m] = f2bf(c[r] * SCALE_);
    }

    // ---- Phase 2: softmax over each row (lane pair m / m+16 splits 2048 cols) ----
    {
        unsigned short* prow = sc + m * ROW_STRIDE + half * 1024;
        float mx = -3.0e38f;
        for (int i = 0; i < 128; ++i) {
            const uint4 d = ((const uint4*)prow)[i];
            mx = fmaxf(mx, fmaxf(bf2f(d.x & 0xFFFFu), bf2f(d.x >> 16)));
            mx = fmaxf(mx, fmaxf(bf2f(d.y & 0xFFFFu), bf2f(d.y >> 16)));
            mx = fmaxf(mx, fmaxf(bf2f(d.z & 0xFFFFu), bf2f(d.z >> 16)));
            mx = fmaxf(mx, fmaxf(bf2f(d.w & 0xFFFFu), bf2f(d.w >> 16)));
        }
        mx = fmaxf(mx, __shfl_xor(mx, 16, 32));

        float sum = 0.f;
        for (int i = 0; i < 128; ++i) {
            uint4 d = ((const uint4*)prow)[i];
            unsigned int* du = (unsigned int*)&d;
            #pragma unroll
            for (int j = 0; j < 4; ++j) {
                const float e0 = __expf(bf2f(du[j] & 0xFFFFu) - mx);
                const float e1 = __expf(bf2f(du[j] >> 16) - mx);
                sum += e0 + e1;
                du[j] = f2bf_pk(e0, e1);
            }
            ((uint4*)prow)[i] = d;
        }
        sum += __shfl_xor(sum, 16, 32);
        if (half == 0) stats[m] = 1.0f / sum;
    }

    // ---- Phase 3: O = P V  (P bf16 from LDS, V^T bf16 from ws) ----
    v8f acc[8];
    #pragma unroll
    for (int nt = 0; nt < 8; ++nt) acc[nt] = (v8f){0.f, 0.f, 0.f, 0.f, 0.f, 0.f, 0.f, 0.f};

    const unsigned short* vtb = vtbf + (size_t)b * E_ * S_;
    for (int kb = 0; kb < 64; ++kb) {
        // A-fragment: lane's 16 k-values live in two contiguous 16B runs -> 2x ds_load_b128
        FragBF pa;
        const unsigned short* abase = sc + m * ROW_STRIDE + kb * 32 + 8 * half;
        pa.q[0] = *(const uint4*)(abase);        // v=0..3 : k = 8*half + 0..7
        pa.q[1] = *(const uint4*)(abase + 16);   // v=4..7 : k = 16 + 8*half + 0..7

        #pragma unroll
        for (int nt = 0; nt < 8; ++nt) {
            FragBF vb;
            const uint4* p = (const uint4*)(vtb + ((size_t)(nt * 16 + m)) * S_ + kb * 32 + 8 * half);
            vb.q[0] = p[0];
            vb.q[1] = p[2];
            acc[nt] = __builtin_amdgcn_wmma_f32_16x16x32_bf16(
                          false, pa.v, false, vb.v, (short)0, acc[nt], false, false);
        }
    }

    // ---- Epilogue: normalize by 1/rowsum, store fp32 ----
    float inv[8];
    #pragma unroll
    for (int r = 0; r < 8; ++r) inv[r] = stats[r + 8 * half];

    float* obase = out + ((size_t)(b * S_ + qb)) * E_;
    #pragma unroll
    for (int nt = 0; nt < 8; ++nt)
        #pragma unroll
        for (int r = 0; r < 8; ++r)
            obase[((size_t)(r + 8 * half)) * E_ + nt * 16 + m] = acc[nt][r] * inv[r];
}

// ---------------------------------------------------------------------------
extern "C" void kernel_launch(void* const* d_in, const int* in_sizes, int n_in,
                              void* d_out, int out_size, void* d_ws, size_t ws_size,
                              hipStream_t stream)
{
    const float* q = (const float*)d_in[0];
    const float* k = (const float*)d_in[1];
    const float* v = (const float*)d_in[2];
    float* out = (float*)d_out;

    const size_t nel = (size_t)B_ * S_ * E_;               // 4,194,304
    unsigned short* kbf  = (unsigned short*)d_ws;          // [B,S,E] bf16
    unsigned short* vtbf = kbf + nel;                      // [B,E,S] bf16 (ws: 16 MB total)

    cvt_kv_kernel<<<(int)(nel / 256), 256, 0, stream>>>(k, v, kbf, vtbf);

    const size_t shmem = 4 * (size_t)REGION;               // 263,424 B (<= 320 KB/WGP)
    attn_kernel<<<(B_ * (S_ / 16)) / 4, 128, shmem, stream>>>(q, kbf, vtbf, out);
}
